// FAdapter_6562710028333
// MI455X (gfx1250) — matrix operand order repaired and checked by reading, not compiled, not run on hardware
//
#include <hip/hip_runtime.h>
#include <hip/hip_bf16.h>
#include <math.h>

typedef __attribute__((ext_vector_type(16))) _Float16 v16h;
typedef __attribute__((ext_vector_type(8)))  _Float16 v8h;
typedef __attribute__((ext_vector_type(8)))  float    v8f;

struct MonaP {
  const float *ln_w, *ln_b, *gamma, *gammax;
  const float *p1_w, *p1_b;
  const float *c1_w, *c1_b, *c2_w, *c2_b, *c3_w, *c3_b;
  const float *bn_w, *bn_b, *proj_w, *proj_b, *p2_w, *p2_b;
};

union AV16 { v16h v; v8h h[2]; };

// A tile 16x32 f16 (M x K), row-major source with 'stride' halves per row.
// ISA layout: lanes 0-15 M=0..15, e=0..7 -> K=hl*8+e ; e=8..15 -> K=16+hl*8+(e-8)
__device__ __forceinline__ v16h ld_a(const _Float16* base, int stride, int mb, int kb, int lane) {
  const int m  = mb + (lane & 15);
  const int hl = lane >> 4;
  const _Float16* p = base + m * stride + kb + hl * 8;
  AV16 u;
  u.h[0] = *(const v8h*)(p);
  u.h[1] = *(const v8h*)(p + 16);
  return u.v;
}

// B tile 32x16 f16 (K x N), stored as [n][k] with 'stride' halves per n-row.
// ISA layout: lane n=lane&15; e=0..7 -> K=hl*8+e ; e=8..15 -> K=16+hl*8+(e-8)
__device__ __forceinline__ v16h ld_b(const _Float16* base, int stride, int nb, int kb, int lane) {
  const int n  = nb + (lane & 15);
  const int hl = lane >> 4;
  const _Float16* p = base + n * stride + kb + hl * 8;
  AV16 u;
  u.h[0] = *(const v8h*)(p);
  u.h[1] = *(const v8h*)(p + 16);
  return u.v;
}

#define WMMA_F16(a, b, c) \
  __builtin_amdgcn_wmma_f32_16x16x32_f16(false, (a), false, (b), (short)0, (c), false, false)

// Full Mona block on one (64 px x 256 ch) patch. Result left in xbuf[j*256+c].
__device__ void mona_patch(const float* __restrict__ xg, int HW, const MonaP& P,
                           float* xbuf, _Float16* abuf, _Float16* wbuf,
                           float* hbuf, _Float16* gbuf,
                           float* red, float* muv, float* rstdv)
{
  const int t    = threadIdx.x;   // 0..127
  const int lane = t & 31;
  const int wave = t >> 5;        // 0..3
  _Float16* wp   = abuf;                       // proj weights reuse xa space (post-GEMM1)
  float*    ybuf = (float*)(abuf + 4096);      // y f32 (16KB) in dead xa space

  // Stage x patch (coalesced over j) and p1 weights (64x256 row-major -> [o][c] = B^T)
  for (int idx = t; idx < 64 * 256; idx += 128) {
    int c = idx >> 6, j = idx & 63;
    xbuf[j * 256 + c] = xg[(size_t)c * HW + j];
  }
  for (int idx = t; idx < 64 * 256; idx += 128)
    wbuf[idx] = (_Float16)P.p1_w[idx];
  __syncthreads();

  // LayerNorm stats: 2 threads per pixel row over 256 channels
  {
    int j = t >> 1, half = t & 1;
    const float* row = xbuf + j * 256 + half * 128;
    float s = 0.f, q = 0.f;
    for (int c = 0; c < 128; ++c) { float v = row[c]; s += v; q += v * v; }
    red[t] = s; red[128 + t] = q;
  }
  __syncthreads();
  if (t < 64) {
    float s = red[2 * t] + red[2 * t + 1];
    float q = red[128 + 2 * t] + red[128 + 2 * t + 1];
    float m = s * (1.f / 256.f);
    float v = q * (1.f / 256.f) - m * m;
    muv[t]   = m;
    rstdv[t] = rsqrtf(v + 1e-5f);
  }
  if (t == 64) {            // NAM normalizer: 1/sum|bn_w|
    float sa = 0.f;
    for (int c = 0; c < 64; ++c) sa += fabsf(P.bn_w[c]);
    red[256] = 1.f / sa;
  }
  __syncthreads();
  const float inv_sabs = red[256];

  // xa = ln*gamma + x*gammax  -> f16 A operand
  for (int idx = t; idx < 64 * 256; idx += 128) {
    int j = idx >> 8, c = idx & 255;
    float x  = xbuf[idx];
    float ln = (x - muv[j]) * rstdv[j] * P.ln_w[c] + P.ln_b[c];
    abuf[idx] = (_Float16)(ln * P.gamma[c] + x * P.gammax[c]);
  }
  __syncthreads();

  // GEMM1: h(64x64) = xa(64x256) @ p1^T + b1 ; wave w -> M-tile w, N-tiles 0..3, K=256
  {
    const int mb = wave * 16;
    v8f acc[4];
#pragma unroll
    for (int nt = 0; nt < 4; ++nt) {
      float bv = P.p1_b[nt * 16 + (lane & 15)];
#pragma unroll
      for (int r = 0; r < 8; ++r) acc[nt][r] = bv;
    }
    for (int kt = 0; kt < 8; ++kt) {
      v16h a = ld_a(abuf, 256, mb, kt * 32, lane);
#pragma unroll
      for (int nt = 0; nt < 4; ++nt) {
        v16h b = ld_b(wbuf, 256, nt * 16, kt * 32, lane);
        acc[nt] = WMMA_F16(a, b, acc[nt]);
      }
    }
#pragma unroll
    for (int nt = 0; nt < 4; ++nt)
#pragma unroll
      for (int r = 0; r < 8; ++r)
        hbuf[(mb + r + ((lane >> 4) << 3)) * 64 + nt * 16 + (lane & 15)] = acc[nt][r];
  }
  __syncthreads();

  // Stage p2 (256x64 row-major = B^T) and proj (64x64 row-major = B^T) while doing convs
  for (int idx = t; idx < 64 * 256; idx += 128)
    wbuf[idx] = (_Float16)P.p2_w[idx];
  for (int idx = t; idx < 64 * 64; idx += 128)
    wp[idx] = (_Float16)P.proj_w[idx];

  // MonaOp: 3 depthwise convs (SAME, on the 8x8 view) + NAM gate; y -> ybuf (f32)
  for (int idx = t; idx < 64 * 64; idx += 128) {
    int ch = idx >> 6;
    int p  = idx & 63;
    int py = p >> 3, px = p & 7;
    float h0 = hbuf[p * 64 + ch];
    float s1 = P.c1_b[ch];
    for (int dy = -1; dy <= 1; ++dy)
      for (int dx = -1; dx <= 1; ++dx) {
        int yy = py + dy, xx = px + dx;
        if ((unsigned)yy < 8u && (unsigned)xx < 8u)
          s1 += hbuf[(yy * 8 + xx) * 64 + ch] * P.c1_w[ch * 9 + (dy + 1) * 3 + (dx + 1)];
      }
    float s2 = P.c2_b[ch];
    for (int dy = -2; dy <= 2; ++dy)
      for (int dx = -2; dx <= 2; ++dx) {
        int yy = py + dy, xx = px + dx;
        if ((unsigned)yy < 8u && (unsigned)xx < 8u)
          s2 += hbuf[(yy * 8 + xx) * 64 + ch] * P.c2_w[ch * 25 + (dy + 2) * 5 + (dx + 2)];
      }
    float s3 = P.c3_b[ch];
    for (int dy = -3; dy <= 3; ++dy)
      for (int dx = -3; dx <= 3; ++dx) {
        int yy = py + dy, xx = px + dx;
        if ((unsigned)yy < 8u && (unsigned)xx < 8u)
          s3 += hbuf[(yy * 8 + xx) * 64 + ch] * P.c3_w[ch * 49 + (dy + 3) * 7 + (dx + 3)];
      }
    float bw  = P.bn_w[ch];
    float xb  = h0 * bw + P.bn_b[ch];
    float wbn = fabsf(bw) * inv_sabs;
    float c4  = h0 / (1.f + expf(-xb * wbn));
    float yv  = (s1 + s2 + s3 + c4) * 0.25f + h0;
    ybuf[p * 64 + ch] = yv;
    gbuf[p * 64 + ch] = (_Float16)yv;   // A operand for proj GEMM
  }
  __syncthreads();

  // GEMM2: P(64x64) = y @ proj^T + proj_b ; then y += P ; g = gelu(y) (exact erf)
  {
    const int mb = wave * 16;
    v8f acc[4];
#pragma unroll
    for (int nt = 0; nt < 4; ++nt) {
      float bv = P.proj_b[nt * 16 + (lane & 15)];
#pragma unroll
      for (int r = 0; r < 8; ++r) acc[nt][r] = bv;
    }
    for (int kt = 0; kt < 2; ++kt) {
      v16h a = ld_a(gbuf, 64, mb, kt * 32, lane);
#pragma unroll
      for (int nt = 0; nt < 4; ++nt) {
        v16h b = ld_b(wp, 64, nt * 16, kt * 32, lane);
        acc[nt] = WMMA_F16(a, b, acc[nt]);
      }
    }
    __syncthreads();   // all waves done reading gbuf before it is overwritten with g
#pragma unroll
    for (int nt = 0; nt < 4; ++nt)
#pragma unroll
      for (int r = 0; r < 8; ++r) {
        int m = mb + r + ((lane >> 4) << 3);
        int n = nt * 16 + (lane & 15);
        float yv = ybuf[m * 64 + n] + acc[nt][r];
        float g  = 0.5f * yv * (1.f + erff(yv * 0.70710678118654752f));
        gbuf[m * 64 + n] = (_Float16)g;
      }
  }
  __syncthreads();

  // GEMM3: out(64x256) = x + g(64x64) @ p2^T + p2_b ; written back into xbuf in place
  {
    const int mb = wave * 16;
    for (int ng = 0; ng < 4; ++ng) {
      v8f acc[4];
#pragma unroll
      for (int q = 0; q < 4; ++q) {
        int n = (ng * 4 + q) * 16 + (lane & 15);
        float bv = P.p2_b[n];
#pragma unroll
        for (int r = 0; r < 8; ++r)
          acc[q][r] = xbuf[(mb + r + ((lane >> 4) << 3)) * 256 + n] + bv;
      }
      for (int kt = 0; kt < 2; ++kt) {
        v16h a = ld_a(gbuf, 64, mb, kt * 32, lane);
#pragma unroll
        for (int q = 0; q < 4; ++q) {
          v16h b = ld_b(wbuf, 64, (ng * 4 + q) * 16, kt * 32, lane);
          acc[q] = WMMA_F16(a, b, acc[q]);
        }
      }
#pragma unroll
      for (int q = 0; q < 4; ++q)
#pragma unroll
        for (int r = 0; r < 8; ++r)
          xbuf[(mb + r + ((lane >> 4) << 3)) * 256 + (ng * 4 + q) * 16 + (lane & 15)] = acc[q][r];
    }
  }
  __syncthreads();
}

#define MONA_SHARED                                        \
  __shared__ alignas(16) float     xbuf[64 * 256];         \
  __shared__ alignas(16) _Float16  abuf[64 * 256];         \
  __shared__ alignas(16) _Float16  wbuf[64 * 256];         \
  __shared__ alignas(16) float     hbuf[64 * 64];          \
  __shared__ alignas(16) _Float16  gbuf[64 * 64];          \
  __shared__ alignas(16) float     red[260];               \
  __shared__ alignas(16) float     muv[64];                \
  __shared__ alignas(16) float     rstdv[64];

// Kernel 1: x1 patches -> y1 written directly to d_out (fully initializes it)
__global__ void __launch_bounds__(128)
k_mona_x1(const float* __restrict__ x, float* __restrict__ out, MonaP P)
{
  MONA_SHARED
  const int pid = blockIdx.x;          // 1024 blocks: 4 batches x 256 patches
  const int b = pid >> 8, s = pid & 255;
  const int HW = 128 * 128;
  const float* xg = x + (size_t)b * 256 * HW + (size_t)s * 64;

  mona_patch(xg, HW, P, xbuf, abuf, wbuf, hbuf, gbuf, red, muv, rstdv);

  float* og = out + (size_t)b * 256 * HW + (size_t)s * 64;
  for (int idx = threadIdx.x; idx < 64 * 256; idx += 128) {
    int c = idx >> 6, j = idx & 63;
    og[(size_t)c * HW + j] = xbuf[j * 256 + c];   // coalesced over j
  }
}

// Kernel 2: one vertical pair of x2 patches per block; 2x2 max-pool block-local,
// then out += pooled (disjoint regions per block; out already holds y1)
__global__ void __launch_bounds__(128)
k_mona_x2_pool(const float* __restrict__ x, float* __restrict__ out, MonaP P)
{
  MONA_SHARED
  const int t   = threadIdx.x;
  const int pid = blockIdx.x;          // 2048 blocks: 4 batches x 128 row-pairs x 4 col-groups
  const int b = pid >> 9, rem = pid & 511;
  const int rp = rem >> 2, cg = rem & 3;     // pooled row rp, cols [cg*32, cg*32+32)
  const int s0 = (2 * rp) * 4 + cg;          // top patch; bottom patch = s0 + 4
  const int HW = 256 * 256;
  const float* xg0 = x + (size_t)b * 256 * HW + (size_t)s0 * 64;

  mona_patch(xg0, HW, P, xbuf, abuf, wbuf, hbuf, gbuf, red, muv, rstdv);

  float pm[64];                               // horizontal pair-max of top patch, in regs
#pragma unroll
  for (int i = 0; i < 64; ++i) {
    int idx = t + i * 128;                    // 8192 = 256 ch x 32 cols
    int c = idx >> 5, wl = idx & 31;
    pm[i] = fmaxf(xbuf[(2 * wl) * 256 + c], xbuf[(2 * wl + 1) * 256 + c]);
  }
  __syncthreads();

  mona_patch(xg0 + 256, HW, P, xbuf, abuf, wbuf, hbuf, gbuf, red, muv, rstdv);

  const int HWo = 128 * 128;
  float* og = out + (size_t)b * 256 * HWo + (size_t)rp * 128 + (size_t)cg * 32;
#pragma unroll
  for (int i = 0; i < 64; ++i) {
    int idx = t + i * 128;
    int c = idx >> 5, wl = idx & 31;
    float m = fmaxf(pm[i], fmaxf(xbuf[(2 * wl) * 256 + c], xbuf[(2 * wl + 1) * 256 + c]));
    float* p = og + (size_t)c * HWo + wl;
    *p = *p + m;
  }
}

extern "C" void kernel_launch(void* const* d_in, const int* in_sizes, int n_in,
                              void* d_out, int out_size, void* d_ws, size_t ws_size,
                              hipStream_t stream) {
  (void)in_sizes; (void)n_in; (void)out_size; (void)d_ws; (void)ws_size;
  const float* x1 = (const float*)d_in[0];
  const float* x2 = (const float*)d_in[1];

  auto mk = [&](int base) {
    MonaP P;
    P.ln_w   = (const float*)d_in[base + 0];
    P.ln_b   = (const float*)d_in[base + 1];
    P.gamma  = (const float*)d_in[base + 2];
    P.gammax = (const float*)d_in[base + 3];
    P.p1_w   = (const float*)d_in[base + 4];
    P.p1_b   = (const float*)d_in[base + 5];
    P.c1_w   = (const float*)d_in[base + 6];
    P.c1_b   = (const float*)d_in[base + 7];
    P.c2_w   = (const float*)d_in[base + 8];
    P.c2_b   = (const float*)d_in[base + 9];
    P.c3_w   = (const float*)d_in[base + 10];
    P.c3_b   = (const float*)d_in[base + 11];
    P.bn_w   = (const float*)d_in[base + 12];
    P.bn_b   = (const float*)d_in[base + 13];
    P.proj_w = (const float*)d_in[base + 14];
    P.proj_b = (const float*)d_in[base + 15];
    P.p2_w   = (const float*)d_in[base + 16];
    P.p2_b   = (const float*)d_in[base + 17];
    return P;
  };
  MonaP P1 = mk(2);
  MonaP P2 = mk(20);

  k_mona_x1<<<dim3(1024), dim3(128), 0, stream>>>(x1, (float*)d_out, P1);
  k_mona_x2_pool<<<dim3(2048), dim3(128), 0, stream>>>(x2, (float*)d_out, P2);
}
